// ContextParallelAttention_17368847745270
// MI455X (gfx1250) — matrix operand order repaired and checked by reading, not compiled, not run on hardware
//
#include <hip/hip_runtime.h>

typedef __attribute__((ext_vector_type(16))) _Float16 v16h;
typedef __attribute__((ext_vector_type(8)))  _Float16 v8h;
typedef __attribute__((ext_vector_type(8)))  float    v8f;
typedef __attribute__((ext_vector_type(4)))  unsigned int u32x4;
typedef __attribute__((ext_vector_type(8)))  int          i32x8;
typedef __attribute__((ext_vector_type(4)))  int          i32x4;

// Problem constants: B=2, H=16, S=2048, D=128
constexpr int Bc    = 2;
constexpr int Hc    = 16;
constexpr int Sc    = 2048;
constexpr int Dc    = 128;
constexpr int QTILE = 128;       // q rows per block (8 waves x 16 rows)
constexpr int KTILE = 32;        // keys per chunk
constexpr int RSTR  = Dc + 4;    // raw fp32 LDS row stride (TDM pads 4 dwords / 128)
constexpr int PSTR  = 40;        // P-bounce row stride (halves), keeps 16B alignment

__device__ inline v8h cvt8(float4 a, float4 b) {
    v8h r;
    r[0] = (_Float16)a.x; r[1] = (_Float16)a.y; r[2] = (_Float16)a.z; r[3] = (_Float16)a.w;
    r[4] = (_Float16)b.x; r[5] = (_Float16)b.y; r[6] = (_Float16)b.z; r[7] = (_Float16)b.w;
    return r;
}

__device__ inline v16h cat16(v8h a, v8h b) {
    v16h r;
#pragma unroll
    for (int i = 0; i < 8; ++i) { r[i] = a[i]; r[8 + i] = b[i]; }
    return r;
}

__device__ inline v8f wmma_f16(v16h a, v16h b, v8f c) {
    return __builtin_amdgcn_wmma_f32_16x16x32_f16(false, a, false, b, (short)0, c, false, false);
}

// TDM: async DMA of a 32x128 fp32 tile (row-major, row stride 128) from global
// into LDS, with 4-dword padding inserted every 128 dwords (-> 132-dword rows).
__device__ inline void tdm_load_tile_32x128(const float* gsrc, unsigned lds_off) {
    unsigned long long ga = (unsigned long long)(uintptr_t)gsrc;
    u32x4 g0 = { 1u,                                            // count = 1 valid descriptor
                 lds_off,                                       // lds_addr (bytes)
                 (unsigned)(ga & 0xffffffffu),                  // global_addr[31:0]
                 (unsigned)((ga >> 32) & 0x1ffffffu) | 0x80000000u }; // ga[56:32] | type=2
    i32x8 g1 = { (int)((2u << 16)          // data_size = 4B
                     | (1u << 20)          // pad_enable
                     | (6u << 22)          // pad_interval = 128 dwords
                     | (3u << 25)),        // pad_amount   = 4 dwords
                 (int)(128u << 16),        // tensor_dim0 = 128 (elements)
                 (int)(32u  << 16),        // tensor_dim1 = 32  (rows)
                 (int)(128u << 16),        // tile_dim0   = 128
                 32,                       // tile_dim1   = 32
                 128,                      // tensor_dim0_stride = 128
                 0, 0 };
    i32x4 z4 = {0, 0, 0, 0};
#if defined(__clang_major__) && (__clang_major__ >= 23)
    i32x8 z8 = {0, 0, 0, 0, 0, 0, 0, 0};
    __builtin_amdgcn_tensor_load_to_lds(g0, g1, z4, z4, z8, 0);
#else
    __builtin_amdgcn_tensor_load_to_lds(g0, g1, z4, z4, 0);
#endif
}

__global__ __launch_bounds__(256, 1)
void fa_causal_wmma_tdm_kernel(const float* __restrict__ Q,
                               const float* __restrict__ K,
                               const float* __restrict__ V,
                               float*       __restrict__ O)
{
    // Double-buffered raw fp32 K/V tiles, filled by the Tensor Data Mover.
    __shared__ float rawK[2][KTILE][RSTR];
    __shared__ float rawV[2][KTILE][RSTR];
    // Per-wave P^T -> P(A-layout) bounce scratch.
    __shared__ _Float16 Pb[8][16][PSTR];

    const int tid  = threadIdx.x;
    const int lane = tid & 31;
    const int wid  = tid >> 5;

    const int  qt    = blockIdx.x & 15;     // S/QTILE = 16
    const int  bh    = blockIdx.x >> 4;     // b*H + h
    const long gbase = (long)bh * Sc * Dc;
    const int  qrow0 = qt * QTILE + wid * 16;
    const float scale = 0.08838834764831845f; // 1/sqrt(128)

    // Fragment lane geometry:
    const int arow = lane & 15;                       // A-layout: lane -> row
    const int aoff = (lane < 16) ? 0 : 8;             // A-layout K offset
    const int bsel = (lane & 15) + ((lane >> 4) ? 16 : 0); // unused; B-layout maps lane -> N column
    const int h8   = (lane >> 4) << 3;                // C-layout row offset of hi lanes

    // ---- Q^T B-fragments, built once (lane = q row, halves = 16 dims), scaled.
    v16h qb[4];
#pragma unroll
    for (int kc = 0; kc < 4; ++kc) {
        const int dim = kc * 32 + aoff;
        const float* qp = Q + gbase + (long)(qrow0 + arow) * Dc + dim;
        v16h f;
#pragma unroll
        for (int i = 0; i < 16; ++i) f[i] = (_Float16)(qp[i < 8 ? i : i + 8] * scale);
        qb[kc] = f;
    }

    const v8f vzero = {0.f, 0.f, 0.f, 0.f, 0.f, 0.f, 0.f, 0.f};
    v8f acc[8];
#pragma unroll
    for (int n = 0; n < 8; ++n) acc[n] = vzero;
    float mst = -1e30f, lst = 0.f;   // per-lane softmax state (one q row per lane)

    const int nch = ((qt * QTILE + QTILE - 1) >> 5) + 1;  // causal chunk count

    // LDS byte offsets of the raw buffers (flat address truncates to LDS offset).
    const unsigned kofs[2] = { (unsigned)(uintptr_t)&rawK[0][0][0],
                               (unsigned)(uintptr_t)&rawK[1][0][0] };
    const unsigned vofs[2] = { (unsigned)(uintptr_t)&rawV[0][0][0],
                               (unsigned)(uintptr_t)&rawV[1][0][0] };

    // ---- TDM prologue: stage chunk 0 into buffer 0 (wave 0 only; EXEC ignored).
    if (wid == 0) {
        tdm_load_tile_32x128(K + gbase, kofs[0]);
        tdm_load_tile_32x128(V + gbase, vofs[0]);
    }

    for (int j = 0; j < nch; ++j) {
        const int jb  = j * KTILE;
        const int cur = j & 1;

        // ---- issue next chunk's DMA, then wait for current chunk (in-order TDM).
        if (wid == 0) {
            if (j + 1 < nch) {
                const long nb = gbase + (long)(jb + KTILE) * Dc;
                tdm_load_tile_32x128(K + nb, kofs[cur ^ 1]);
                tdm_load_tile_32x128(V + nb, vofs[cur ^ 1]);
                __builtin_amdgcn_s_wait_tensorcnt((short)2);  // oldest 2 (chunk j) done
            } else {
                __builtin_amdgcn_s_wait_tensorcnt((short)0);
            }
        }
        __syncthreads();

        // waves fully above the diagonal skip compute (uniform per wave)
        if (jb <= qrow0 + 15) {
            const float (*rk)[RSTR] = rawK[cur];
            const float (*rv)[RSTR] = rawV[cur];

            // ---- S^T = K Q^T : two 16(key)x16(qrow) tiles, contraction D=128.
            v8f st0 = vzero, st1 = vzero;
#pragma unroll
            for (int kc = 0; kc < 4; ++kc) {
                const int d0 = kc * 32 + aoff;
                const float4* a00 = (const float4*)&rk[arow][d0];
                const float4* a01 = (const float4*)&rk[arow][d0 + 16];
                v16h ak0 = cat16(cvt8(a00[0], a00[1]), cvt8(a01[0], a01[1]));
                const float4* a10 = (const float4*)&rk[16 + arow][d0];
                const float4* a11 = (const float4*)&rk[16 + arow][d0 + 16];
                v16h ak1 = cat16(cvt8(a10[0], a10[1]), cvt8(a11[0], a11[1]));
                st0 = wmma_f16(ak0, qb[kc], st0);
                st1 = wmma_f16(ak1, qb[kc], st1);
            }

            // ---- causal mask: element key = jb + t*16 + r + h8, qrow = qrow0 + arow
            if (jb + 31 > qrow0) {
                const int qr = qrow0 + arow;
#pragma unroll
                for (int r = 0; r < 8; ++r) {
                    if (jb + r + h8      > qr) st0[r] = -1e30f;
                    if (jb + 16 + r + h8 > qr) st1[r] = -1e30f;
                }
            }

            // ---- online softmax: each lane holds 16 of the 32 keys of its q row.
            float mx = st0[0];
#pragma unroll
            for (int r = 1; r < 8; ++r) mx = fmaxf(mx, st0[r]);
#pragma unroll
            for (int r = 0; r < 8; ++r) mx = fmaxf(mx, st1[r]);
            mx = fmaxf(mx, __shfl_xor(mx, 16, 32));   // combine the two key halves

            const float mn = fmaxf(mst, mx);
            const float al = __expf(mst - mn);
            float rs = 0.f;
#pragma unroll
            for (int r = 0; r < 8; ++r) {
                st0[r] = __expf(st0[r] - mn); rs += st0[r];
                st1[r] = __expf(st1[r] - mn); rs += st1[r];
            }
            rs  = rs + __shfl_xor(rs, 16, 32);
            lst = lst * al + rs;
            mst = mn;

            // rescale accumulators: alpha indexed by q row = r + h8 -> lane r + h8
#pragma unroll
            for (int r = 0; r < 8; ++r) {
                const float alr = __shfl(al, r + h8, 32);
#pragma unroll
                for (int n = 0; n < 8; ++n) acc[n][r] *= alr;
            }

            // ---- P^T (C-layout) -> P (A-layout) via per-wave LDS bounce.
#pragma unroll
            for (int r = 0; r < 8; ++r) {
                Pb[wid][arow][h8 + r]      = (_Float16)st0[r];
                Pb[wid][arow][16 + h8 + r] = (_Float16)st1[r];
            }
            asm volatile("s_wait_dscnt 0x0" ::: "memory");  // wave-private RAW on LDS
            v16h pa = cat16(*(const v8h*)&Pb[wid][arow][aoff],
                            *(const v8h*)&Pb[wid][arow][aoff + 16]);

            // ---- O += P V : V B-fragments straight from raw fp32 tile (lane = dim).
#pragma unroll
            for (int n = 0; n < 8; ++n) {
                v16h vb;
#pragma unroll
                for (int s = 0; s < 16; ++s) vb[s] = (_Float16)rv[s < 8 ? aoff + s : 16 + aoff + (s - 8)][n * 16 + arow];
                acc[n] = wmma_f16(pa, vb, acc[n]);
            }
        }
        __syncthreads();
    }

    // ---- normalize and store: 1/l indexed by q row = r + h8 -> lane r + h8.
    const float linv = 1.0f / lst;
#pragma unroll
    for (int r = 0; r < 8; ++r) {
        const float inv = __shfl(linv, r + h8, 32);
        const int   row = qrow0 + h8 + r;
        float* op = O + gbase + (long)row * Dc + arow;
#pragma unroll
        for (int n = 0; n < 8; ++n) op[n * 16] = acc[n][r] * inv;
    }
}

extern "C" void kernel_launch(void* const* d_in, const int* in_sizes, int n_in,
                              void* d_out, int out_size, void* d_ws, size_t ws_size,
                              hipStream_t stream) {
    const float* q = (const float*)d_in[0];
    const float* k = (const float*)d_in[1];
    const float* v = (const float*)d_in[2];
    float* out = (float*)d_out;

    dim3 grid(Bc * Hc * (Sc / QTILE));  // 512 blocks
    dim3 block(256);                    // 8 waves (wave32)
    fa_causal_wmma_tdm_kernel<<<grid, block, 0, stream>>>(q, k, v, out);
}
